// MultiHeadAttention_75608604279044
// MI455X (gfx1250) — compile-verified, matrix operand order checked by reading, not run on hardware
//
#include <hip/hip_runtime.h>
#include <hip/hip_bf16.h>

typedef __attribute__((ext_vector_type(16))) _Float16 v16h;
typedef __attribute__((ext_vector_type(8)))  _Float16 v8h;
typedef __attribute__((ext_vector_type(4)))  _Float16 v4h;
typedef __attribute__((ext_vector_type(8)))  float    v8f;
typedef __attribute__((ext_vector_type(4)))  float    v4f;

#define D_MODEL 1024
#define SEQ     2048
#define NHEAD   16
#define HDIM    64
#define BATCH   2
#define NTOK    (BATCH * SEQ)   // 4096

// --- gfx1250 async global->LDS path (guarded; falls back to sync staging) ---
#if defined(__has_builtin)
#  if __has_builtin(__builtin_amdgcn_global_load_async_to_lds_b128) && \
      __has_builtin(__builtin_amdgcn_s_wait_asynccnt)
#    define USE_ASYNC_LDS 1
#  endif
#endif
#ifndef USE_ASYNC_LDS
#  define USE_ASYNC_LDS 0
#endif

#if USE_ASYNC_LDS
// Signature (from hipcc diagnostic): param0 = global (AS1) pointer to
// GCC-vector int4, param1 = LDS (AS3) pointer to int4, then imm offset, cpol.
typedef __attribute__((__vector_size__(4 * sizeof(int)))) int gv4i;
typedef __attribute__((address_space(1))) gv4i* g_as1_v4i;
typedef __attribute__((address_space(3))) gv4i* l_as3_v4i;

__device__ __forceinline__ void async_b128(const _Float16* gsrc, _Float16* ldst) {
  __builtin_amdgcn_global_load_async_to_lds_b128(
      (g_as1_v4i)gsrc, (l_as3_v4i)ldst, 0, 0);
}
#endif

// ---------------------------------------------------------------------------
// WMMA helpers (CDNA5: D = A(16x32 f16) * B(32x16 f16) + C(16x16 f32))
// ---------------------------------------------------------------------------
__device__ __forceinline__ v8f wmma16(v16h a, v16h b, v8f c) {
  return __builtin_amdgcn_wmma_f32_16x16x32_f16(
      /*neg_a=*/false, a, /*neg_b=*/false, b,
      /*c_mod=*/(short)0, c, /*reuse_a=*/false, /*reuse_b=*/false);
}

// Per-lane fragment load for BOTH A and B operands of 16x16x32:
// lane holds one "row" (A: M-row, B: N-column of W stored row-major [N,K]);
// 16 f16 elements = two contiguous 8-half chunks at k0=(lane>>4)*8 and k0+16.
__device__ __forceinline__ v16h frag_from_row(const _Float16* row, int kbase, int lane) {
  int k0 = kbase + ((lane >> 4) << 3);
  v8h lo = *(const v8h*)(row + k0);
  v8h hi = *(const v8h*)(row + k0 + 16);
  v16h out;
#pragma unroll
  for (int i = 0; i < 8; ++i) { out[i] = lo[i]; out[i + 8] = hi[i]; }
  return out;
}

// ---------------------------------------------------------------------------
// fp32 -> fp16 conversion (vectorized, grid-stride)
// ---------------------------------------------------------------------------
__global__ void cvt_f32_to_f16(const float* __restrict__ src,
                               _Float16* __restrict__ dst, int n4) {
  int i = blockIdx.x * blockDim.x + threadIdx.x;
  int stride = gridDim.x * blockDim.x;
  for (; i < n4; i += stride) {
    v4f v = ((const v4f*)src)[i];
    v4h o;
    o[0] = (_Float16)v[0]; o[1] = (_Float16)v[1];
    o[2] = (_Float16)v[2]; o[3] = (_Float16)v[3];
    ((v4h*)dst)[i] = o;
  }
}

// ---------------------------------------------------------------------------
// QKV projection: Y = X @ W^T  (X:[NTOK,D], W:[D,D] torch-style [out,in])
// Block = 256 threads (8 waves), tile 128(M) x 64(N), K-step 32.
// Double-buffered B tile in LDS; async global->LDS DMA overlapped with WMMA.
// mode 0/1 -> write [B,H,S,hd] (Q,K); mode 2 -> write [B,H,hd,S] (V^T).
// ---------------------------------------------------------------------------
__global__ void qkv_gemm(const _Float16* __restrict__ X,
                         const _Float16* __restrict__ W,
                         _Float16* __restrict__ Out, int mode) {
  __shared__ __align__(16) _Float16 Bs[2][64 * 32];
  const int lane = threadIdx.x & 31;
  const int wave = threadIdx.x >> 5;     // 0..7
  const int half = lane >> 4;
  const int l16  = lane & 15;
  const int Mbase = blockIdx.y * 128;
  const int Nbase = blockIdx.x * 64;
  // cooperative B-tile mapping: 2048 halves / 256 threads = one b128 each
  const int ln = threadIdx.x >> 2;        // 0..63 (W row within tile)
  const int lk = (threadIdx.x & 3) * 8;   // 0,8,16,24 (k within tile)

  v8f acc[4] = {};
  const _Float16* Arow = X + (size_t)(Mbase + wave * 16 + l16) * D_MODEL;
  const _Float16* Bsrc = W + (size_t)(Nbase + ln) * D_MODEL + lk;
  _Float16* Bdst[2] = { &Bs[0][ln * 32 + lk], &Bs[1][ln * 32 + lk] };

  // preload tile 0 into buffer 0
#if USE_ASYNC_LDS
  async_b128(Bsrc, Bdst[0]);
#else
  *(v8h*)Bdst[0] = *(const v8h*)Bsrc;
#endif

  int cur = 0;
  for (int kb = 0; kb < D_MODEL; kb += 32, cur ^= 1) {
#if USE_ASYNC_LDS
    __builtin_amdgcn_s_wait_asynccnt(0);   // own DMA for `cur` complete
#endif
    __syncthreads();                       // all waves' tile ready; prior reads of
                                           // the other buffer are also fenced
    if (kb + 32 < D_MODEL) {               // kick DMA for next tile (overlaps WMMA)
#if USE_ASYNC_LDS
      async_b128(Bsrc + kb + 32, Bdst[cur ^ 1]);
#else
      *(v8h*)Bdst[cur ^ 1] = *(const v8h*)(Bsrc + kb + 32);
      __builtin_prefetch(Bsrc + kb + 64, 0, 3);
#endif
    }
    v16h a = frag_from_row(Arow, kb, lane);
    const _Float16* bbase = &Bs[cur][0];
#pragma unroll
    for (int s = 0; s < 4; ++s) {
      v16h b = frag_from_row(bbase + (s * 16 + l16) * 32, 0, lane);
      acc[s] = wmma16(a, b, acc[s]);
    }
  }

#pragma unroll
  for (int s = 0; s < 4; ++s) {
#pragma unroll
    for (int r = 0; r < 8; ++r) {
      int gm = Mbase + wave * 16 + r + 8 * half;  // token index
      int gn = Nbase + s * 16 + l16;              // output feature
      int b  = gm >> 11;                          // /SEQ
      int si = gm & (SEQ - 1);
      int h  = gn >> 6;                           // /HDIM
      int d  = gn & (HDIM - 1);
      _Float16 val = (_Float16)acc[s][r];
      if (mode == 2)
        Out[((size_t)(b * NHEAD + h) * HDIM + d) * SEQ + si] = val;   // V^T
      else
        Out[((size_t)(b * NHEAD + h) * SEQ + si) * HDIM + d] = val;   // Q, K
    }
  }
}

// ---------------------------------------------------------------------------
// Flash attention (causal). grid = (S/64 q-tiles, B*H). 128 threads (4 waves),
// each wave owns 16 q-rows, streaming softmax over k-tiles of 64.
// ---------------------------------------------------------------------------
__global__ void attn_kernel(const _Float16* __restrict__ Q,   // [B*H,S,hd]
                            const _Float16* __restrict__ K,   // [B*H,S,hd]
                            const _Float16* __restrict__ VT,  // [B*H,hd,S]
                            _Float16* __restrict__ Ctx) {     // [NTOK, D_MODEL]
  __shared__ __align__(16) _Float16 Ps[4][16 * 64];  // per-wave transpose buffer
  const int lane = threadIdx.x & 31;
  const int wave = threadIdx.x >> 5;
  const int half = lane >> 4;
  const int l16  = lane & 15;
  const int qt = blockIdx.x;            // q tile (0..31)
  const int bh = blockIdx.y;            // 0..31
  const int b  = bh >> 4;
  const int h  = bh & 15;

  const _Float16* Qp = Q  + (size_t)bh * SEQ * HDIM;
  const _Float16* Kp = K  + (size_t)bh * SEQ * HDIM;
  const _Float16* Vp = VT + (size_t)bh * HDIM * SEQ;

  const int q0 = qt * 64 + wave * 16;   // first q-row of this wave

  v16h aq0 = frag_from_row(Qp + (size_t)(q0 + l16) * HDIM, 0,  lane);
  v16h aq1 = frag_from_row(Qp + (size_t)(q0 + l16) * HDIM, 32, lane);

  float run_m[8], run_l[8];
#pragma unroll
  for (int r = 0; r < 8; ++r) { run_m[r] = -3.0e38f; run_l[r] = 0.0f; }
  v8f o[4] = {};

  for (int j = 0; j <= qt; ++j) {       // causal tile skipping
    // --- scores: S = q(16x64) @ k^T(64x64) ---
    v8f sc[4] = {};
#pragma unroll
    for (int s = 0; s < 4; ++s) {
      const _Float16* krow = Kp + (size_t)(j * 64 + s * 16 + l16) * HDIM;
      sc[s] = wmma16(aq0, frag_from_row(krow, 0,  lane), sc[s]);
      sc[s] = wmma16(aq1, frag_from_row(krow, 32, lane), sc[s]);
    }
    if (j < qt)  // prefetch next k-tile while softmax runs
      __builtin_prefetch(Kp + (size_t)((j + 1) * 64 + l16) * HDIM, 0, 3);

    // --- scale + causal mask (element (r,lane): m=r+8*half, n=s*16+l16) ---
#pragma unroll
    for (int s = 0; s < 4; ++s)
#pragma unroll
      for (int r = 0; r < 8; ++r) {
        float v  = sc[s][r] * 0.125f;   // 1/sqrt(64)
        int   gq = q0 + r + 8 * half;
        int   gk = j * 64 + s * 16 + l16;
        sc[s][r] = (gk > gq) ? -1.0e30f : v;
      }
    // --- online softmax: rows live in VGPR index r, reduce across 16 lanes ---
    float nm[8], corr[8];
#pragma unroll
    for (int r = 0; r < 8; ++r) {
      float m = fmaxf(fmaxf(sc[0][r], sc[1][r]), fmaxf(sc[2][r], sc[3][r]));
#pragma unroll
      for (int off = 1; off <= 8; off <<= 1)
        m = fmaxf(m, __shfl_xor(m, off));
      nm[r]   = fmaxf(run_m[r], m);
      corr[r] = __expf(run_m[r] - nm[r]);
      run_m[r] = nm[r];
    }
#pragma unroll
    for (int r = 0; r < 8; ++r) {
      float sum = 0.0f;
#pragma unroll
      for (int s = 0; s < 4; ++s) {
        float p = __expf(sc[s][r] - nm[r]);
        sc[s][r] = p;
        sum += p;
      }
#pragma unroll
      for (int off = 1; off <= 8; off <<= 1)
        sum += __shfl_xor(sum, off);
      run_l[r] = run_l[r] * corr[r] + sum;
    }
#pragma unroll
    for (int s = 0; s < 4; ++s)
#pragma unroll
      for (int r = 0; r < 8; ++r)
        o[s][r] *= corr[r];

    // --- transpose P (D-layout -> A-layout) via per-wave LDS buffer.
    // LDS ops are in-order within a wave on CDNA5; buffer is wave-private,
    // so no workgroup barrier is required.
#pragma unroll
    for (int s = 0; s < 4; ++s)
#pragma unroll
      for (int r = 0; r < 8; ++r) {
        int m = r + 8 * half;
        Ps[wave][m * 64 + s * 16 + l16] = (_Float16)sc[s][r];
      }
    v16h pa0 = frag_from_row(&Ps[wave][l16 * 64], 0,  lane);
    v16h pa1 = frag_from_row(&Ps[wave][l16 * 64], 32, lane);

    // --- ctx += P(16x64) @ V(64x64), V read as V^T rows (contiguous in j) ---
#pragma unroll
    for (int s = 0; s < 4; ++s) {
      const _Float16* vrow = Vp + (size_t)(s * 16 + l16) * SEQ + j * 64;
      o[s] = wmma16(pa0, frag_from_row(vrow, 0,  lane), o[s]);
      o[s] = wmma16(pa1, frag_from_row(vrow, 32, lane), o[s]);
    }
  }

  // --- normalize and scatter ctx back to token-major [B,S,H*hd] (f16) ---
#pragma unroll
  for (int r = 0; r < 8; ++r) run_l[r] = 1.0f / run_l[r];
#pragma unroll
  for (int s = 0; s < 4; ++s)
#pragma unroll
    for (int r = 0; r < 8; ++r) {
      int gq  = q0 + r + 8 * half;
      int tok = b * SEQ + gq;
      int d   = s * 16 + l16;
      Ctx[(size_t)tok * D_MODEL + h * HDIM + d] = (_Float16)(o[s][r] * run_l[r]);
    }
}

// ---------------------------------------------------------------------------
// Output projection: out = ctx @ Wo^T + bo, fp32 out. Same tiling as qkv_gemm.
// ---------------------------------------------------------------------------
__global__ void out_gemm(const _Float16* __restrict__ X,
                         const _Float16* __restrict__ W,
                         const float* __restrict__ bias,
                         float* __restrict__ Out) {
  __shared__ __align__(16) _Float16 Bs[2][64 * 32];
  const int lane = threadIdx.x & 31;
  const int wave = threadIdx.x >> 5;     // 0..7
  const int half = lane >> 4;
  const int l16  = lane & 15;
  const int Mbase = blockIdx.y * 128;
  const int Nbase = blockIdx.x * 64;
  const int ln = threadIdx.x >> 2;
  const int lk = (threadIdx.x & 3) * 8;

  v8f acc[4] = {};
  const _Float16* Arow = X + (size_t)(Mbase + wave * 16 + l16) * D_MODEL;
  const _Float16* Bsrc = W + (size_t)(Nbase + ln) * D_MODEL + lk;
  _Float16* Bdst[2] = { &Bs[0][ln * 32 + lk], &Bs[1][ln * 32 + lk] };

#if USE_ASYNC_LDS
  async_b128(Bsrc, Bdst[0]);
#else
  *(v8h*)Bdst[0] = *(const v8h*)Bsrc;
#endif

  int cur = 0;
  for (int kb = 0; kb < D_MODEL; kb += 32, cur ^= 1) {
#if USE_ASYNC_LDS
    __builtin_amdgcn_s_wait_asynccnt(0);
#endif
    __syncthreads();
    if (kb + 32 < D_MODEL) {
#if USE_ASYNC_LDS
      async_b128(Bsrc + kb + 32, Bdst[cur ^ 1]);
#else
      *(v8h*)Bdst[cur ^ 1] = *(const v8h*)(Bsrc + kb + 32);
      __builtin_prefetch(Bsrc + kb + 64, 0, 3);
#endif
    }
    v16h a = frag_from_row(Arow, kb, lane);
    const _Float16* bbase = &Bs[cur][0];
#pragma unroll
    for (int s = 0; s < 4; ++s) {
      v16h bfr = frag_from_row(bbase + (s * 16 + l16) * 32, 0, lane);
      acc[s] = wmma16(a, bfr, acc[s]);
    }
  }

#pragma unroll
  for (int s = 0; s < 4; ++s)
#pragma unroll
    for (int r = 0; r < 8; ++r) {
      int gm = Mbase + wave * 16 + r + 8 * half;
      int gn = Nbase + s * 16 + l16;
      Out[(size_t)gm * D_MODEL + gn] = acc[s][r] + bias[gn];
    }
}

// ---------------------------------------------------------------------------
// Host launcher
// ---------------------------------------------------------------------------
extern "C" void kernel_launch(void* const* d_in, const int* in_sizes, int n_in,
                              void* d_out, int out_size, void* d_ws, size_t ws_size,
                              hipStream_t stream) {
  (void)in_sizes; (void)n_in; (void)out_size; (void)ws_size;
  const float* x  = (const float*)d_in[0];
  const float* Wq = (const float*)d_in[1];
  const float* Wk = (const float*)d_in[2];
  const float* Wv = (const float*)d_in[3];
  const float* Wo = (const float*)d_in[4];
  const float* bo = (const float*)d_in[5];
  float* out = (float*)d_out;

  // workspace layout (bytes)
  char* ws = (char*)d_ws;
  const size_t SZ_X  = (size_t)NTOK * D_MODEL * sizeof(_Float16);        // 8 MiB
  const size_t SZ_W  = (size_t)D_MODEL * D_MODEL * sizeof(_Float16);     // 2 MiB
  const size_t SZ_QK = (size_t)BATCH * NHEAD * SEQ * HDIM * sizeof(_Float16); // 8 MiB
  _Float16* xh   = (_Float16*)(ws);
  _Float16* wqh  = (_Float16*)(ws + SZ_X);
  _Float16* wkh  = (_Float16*)(ws + SZ_X + SZ_W);
  _Float16* wvh  = (_Float16*)(ws + SZ_X + 2 * SZ_W);
  _Float16* woh  = (_Float16*)(ws + SZ_X + 3 * SZ_W);
  _Float16* Qh   = (_Float16*)(ws + SZ_X + 4 * SZ_W);
  _Float16* Kh   = (_Float16*)(ws + SZ_X + 4 * SZ_W + SZ_QK);
  _Float16* VTh  = (_Float16*)(ws + SZ_X + 4 * SZ_W + 2 * SZ_QK);
  _Float16* ctxh = (_Float16*)(ws + SZ_X + 4 * SZ_W + 3 * SZ_QK);

  // 1) convert inputs to f16
  {
    int n4 = (NTOK * D_MODEL) / 4;
    cvt_f32_to_f16<<<(n4 + 255) / 256, 256, 0, stream>>>(x, xh, n4);
    int w4 = (D_MODEL * D_MODEL) / 4;
    cvt_f32_to_f16<<<(w4 + 255) / 256, 256, 0, stream>>>(Wq, wqh, w4);
    cvt_f32_to_f16<<<(w4 + 255) / 256, 256, 0, stream>>>(Wk, wkh, w4);
    cvt_f32_to_f16<<<(w4 + 255) / 256, 256, 0, stream>>>(Wv, wvh, w4);
    cvt_f32_to_f16<<<(w4 + 255) / 256, 256, 0, stream>>>(Wo, woh, w4);
  }

  // 2) QKV projections (V written transposed for the attention B-operand)
  dim3 ggrid(D_MODEL / 64, NTOK / 128);  // (16, 32)
  qkv_gemm<<<ggrid, 256, 0, stream>>>(xh, wqh, Qh,  0);
  qkv_gemm<<<ggrid, 256, 0, stream>>>(xh, wkh, Kh,  1);
  qkv_gemm<<<ggrid, 256, 0, stream>>>(xh, wvh, VTh, 2);

  // 3) causal flash attention
  dim3 agrid(SEQ / 64, BATCH * NHEAD);   // (32, 32)
  attn_kernel<<<agrid, 128, 0, stream>>>(Qh, Kh, VTh, ctxh);

  // 4) output projection + bias (fp32 out)
  out_gemm<<<ggrid, 256, 0, stream>>>(ctxh, woh, bo, out);
}
